// Tacotron2Model_89215060672839
// MI455X (gfx1250) — matrix operand order repaired and checked by reading, not compiled, not run on hardware
//
#include <hip/hip_runtime.h>
#include <stdint.h>

// ---------------- types / helpers ----------------
typedef uint16_t bf16_t; // storage-only bf16
typedef __attribute__((ext_vector_type(16))) __bf16 v16bf;
typedef __attribute__((ext_vector_type(8)))  float  v8f;

union FragU { v16bf v; uint4 q[2]; };

__device__ __forceinline__ uint16_t f2bfu(float f) {
  uint32_t u = __float_as_uint(f);
  return (uint16_t)((u + 0x7fffu + ((u >> 16) & 1u)) >> 16); // RNE
}
__device__ __forceinline__ float bf2f(uint16_t s) {
  return __uint_as_float(((uint32_t)s) << 16);
}
__device__ __forceinline__ float sigmf(float x) { return 1.f / (1.f + __expf(-x)); }
__device__ __forceinline__ void pf(const void* p) { __builtin_prefetch(p, 0, 1); } // global_prefetch_b8

// A-operand (16x32 bf16, M rows striped over lanes 0..15, K halves over lane>>4)
__device__ __forceinline__ v16bf a_frag(const uint32_t* row, int kp0, int hk) {
  const uint32_t* p = row + kp0 + hk * 4;
  FragU f;
  f.q[0] = *(const uint4*)(p);
  f.q[1] = *(const uint4*)(p + 8);
  return f.v;
}
// B-operand (32x16 bf16) for D = X * W^T with W stored [N][K] row-major:
// lane n = lid&15 reads dwords kh*4..+3 and 8+kh*4..+3 of weight row n (same k order as A).
__device__ __forceinline__ v16bf b_frag(const uint32_t* wrow, int kp0, int kh) {
  const uint32_t* p = wrow + kp0 + kh * 4;
  FragU f;
  f.q[0] = *(const uint4*)(p);
  f.q[1] = *(const uint4*)(p + 8);
  return f.v;
}
__device__ __forceinline__ v8f wmma_bf16(v16bf a, v16bf b, v8f c) {
  return __builtin_amdgcn_wmma_f32_16x16x32_bf16(false, a, false, b, (short)0, c, false, false);
}

#define GS_LOOP(i, total) \
  for (long i = (long)blockIdx.x * blockDim.x + threadIdx.x; i < (total); i += (long)gridDim.x * blockDim.x)

// ---------------- prep kernels ----------------
__global__ void k_zero(uint32_t* p, long n) { GS_LOOP(i, n) p[i] = 0u; }

// out[n][0:ka] = wa[n][:], out[n][ka:ka+kb] = wb[n][:] (or 0 if wb null); bf16 convert
__global__ void k_prep_cat(const float* wa, int ka, const float* wb, int kb, bf16_t* out, long N) {
  int kk = ka + kb;
  GS_LOOP(i, N * (long)kk) {
    long n = i / kk; int c = (int)(i % kk);
    float v = (c < ka) ? wa[n * ka + c] : (wb ? wb[n * (long)kb + (c - ka)] : 0.f);
    out[i] = f2bfu(v);
  }
}
// conv weight [O=256][I][5] -> bf16 [O][5*I] with K index = tap*I + i
__global__ void k_prep_conv(const float* w, int I, bf16_t* out) {
  long KI = 5L * I;
  GS_LOOP(i, 256L * KI) {
    long o = i / KI; long r = i % KI; int k = (int)(r / I); int ii = (int)(r % I);
    out[i] = f2bfu(w[((o * I) + ii) * 5 + k]);
  }
}
__global__ void k_addb(const float* a, const float* b, float* o, long n) {
  GS_LOOP(i, n) o[i] = a[i] + b[i];
}

// embedding -> bf16 [B][260][512] with 2 zero pad rows each side
__global__ void k_embed(const int* __restrict__ text, const float* __restrict__ emb,
                        bf16_t* __restrict__ xpad0) {
  int br = blockIdx.x; int b = br / 260; int tr = br % 260;
  bf16_t* dst = xpad0 + ((size_t)b * 260 + tr) * 512;
  if (tr < 2 || tr >= 258) {
    for (int i = threadIdx.x; i < 512; i += 256) dst[i] = 0;
    return;
  }
  int v = text[b * 256 + (tr - 2)];
  const float* src = emb + (size_t)v * 512;
  for (int i = threadIdx.x; i < 512; i += 256) dst[i] = f2bfu(src[i]);
}

// ---------------- conv-as-GEMM (K = 5*Cin, contiguous window in padded buf) ----------------
__global__ __launch_bounds__(256) void k_convgemm(const bf16_t* __restrict__ Apad, int Cin,
                                                  const bf16_t* __restrict__ Wc, const float* __restrict__ bias,
                                                  bf16_t* __restrict__ out, int outTp, int opad) {
  const int K = 5 * Cin, Kd = K >> 1, Cind = Cin >> 1;
  const int lid = threadIdx.x & 31;
  const int w = (blockIdx.x * blockDim.x + threadIdx.x) >> 5;
  const int mt = w & 511, ng = w >> 9;           // 512 m-tiles, 4 n-groups
  const int m0 = mt * 16, n0 = ng * 64;
  const int lm = lid & 15, hk = lid >> 4;
  const int m = m0 + lm, b = m >> 8, t = m & 255;
  const uint32_t* arow = (const uint32_t*)Apad + ((size_t)b * 260 + t) * Cind;
  const uint32_t* Wdw = (const uint32_t*)Wc;
  v8f acc[4] = {};
  for (int k0 = 0; k0 < K; k0 += 32) {
    int kp0 = k0 >> 1;
    v16bf a = a_frag(arow, kp0, hk);
    pf(arow + kp0 + 32); // prefetch next A k-chunk
#pragma unroll
    for (int j = 0; j < 4; ++j) {
      const uint32_t* wrow = Wdw + (size_t)(n0 + j * 16 + lm) * Kd;
      v16bf bb = b_frag(wrow, kp0, hk);
      pf(wrow + kp0 + 32); // prefetch next B k-chunk
      acc[j] = wmma_bf16(a, bb, acc[j]);
    }
  }
#pragma unroll
  for (int j = 0; j < 4; ++j) {
    int n = n0 + j * 16 + lm;
    float bv = bias[n];
#pragma unroll
    for (int v = 0; v < 8; ++v) {
      int r = m0 + v + 8 * hk, rb = r >> 8, rt = r & 255;
      float x = acc[j][v] + bv;
      if (x < 0.f) x = 0.f;
      out[((size_t)rb * outTp + rt + opad) * 256 + n] = f2bfu(x);
    }
  }
}

// ---------------- plain GEMM (prenet): out = relu(A*W^T + b), bf16 out ----------------
__global__ __launch_bounds__(256) void k_gemm_lin(const bf16_t* __restrict__ A, const bf16_t* __restrict__ W,
                                                  const float* __restrict__ bias, bf16_t* __restrict__ out,
                                                  int M, int N, int K) {
  const int lid = threadIdx.x & 31;
  const int w = (blockIdx.x * blockDim.x + threadIdx.x) >> 5;
  const int mtiles = M >> 4;
  const int mt = w % mtiles, ng = w / mtiles;
  const int m0 = mt * 16, n0 = ng * 64;
  const int lm = lid & 15, hk = lid >> 4;
  const int Kd = K >> 1;
  const uint32_t* arow = (const uint32_t*)A + (size_t)(m0 + lm) * Kd;
  const uint32_t* Wdw = (const uint32_t*)W;
  v8f acc[4] = {};
  for (int k0 = 0; k0 < K; k0 += 32) {
    int kp0 = k0 >> 1;
    v16bf a = a_frag(arow, kp0, hk);
#pragma unroll
    for (int j = 0; j < 4; ++j) {
      v16bf bb = b_frag(Wdw + (size_t)(n0 + j * 16 + lm) * Kd, kp0, hk);
      acc[j] = wmma_bf16(a, bb, acc[j]);
    }
  }
#pragma unroll
  for (int j = 0; j < 4; ++j) {
    int n = n0 + j * 16 + lm;
    float bv = bias[n];
#pragma unroll
    for (int v = 0; v < 8; ++v) {
      int r = m0 + v + 8 * hk;
      float x = acc[j][v] + bv;
      if (x < 0.f) x = 0.f;
      out[(size_t)r * N + n] = f2bfu(x);
    }
  }
}

// ---------------- encoder bi-LSTM layer (persistent, 1 block per direction) ----------------
// gates = [x_t | h] @ Wcat^T (Wcat [512][384] bf16), staged in LDS; 64KB static LDS total.
__global__ __launch_bounds__(512) void k_enc(const bf16_t* __restrict__ X, int xTB,
                                             const bf16_t* __restrict__ Wf, const float* __restrict__ Bf,
                                             const bf16_t* __restrict__ Wb, const float* __restrict__ Bb,
                                             bf16_t* __restrict__ hsOut, float* __restrict__ encOut) {
  __shared__ __align__(16) uint32_t smem0[8192]; // A(32x384 bf16, 192dw/row) or gates(32x512 bf16)
  __shared__ float hbuf[4096];
  __shared__ float cbuf[4096];
  const int tid = threadIdx.x;
  const int dir = blockIdx.x;
  const uint32_t* Wdw = (const uint32_t*)(dir ? Wb : Wf); // rows 512, Kd=192
  const float* Bs = dir ? Bb : Bf;
  const uint32_t* Xdw = (const uint32_t*)X; // 128 dwords per (t,b) row
  for (int i = tid; i < 4096; i += 512) { hbuf[i] = 0.f; cbuf[i] = 0.f; }
  __syncthreads();
  const int lid = tid & 31, w = tid >> 5;
  const int lm = lid & 15, hk = lid >> 4;
  const int n0 = w * 32;
  for (int t = 0; t < 256; ++t) {
    int tt = dir ? (255 - t) : t;
    for (int e = tid; e < 4096; e += 512) { // stage x_t (32x256 bf16)
      int b = e >> 7, d = e & 127;
      size_t src = xTB ? ((size_t)(tt * 32 + b) * 128 + d) : ((size_t)(b * 256 + tt) * 128 + d);
      smem0[b * 192 + d] = Xdw[src];
    }
    for (int e = tid; e < 2048; e += 512) { // stage h (32x128 f32 -> bf16 pairs)
      int b = e >> 6, d = e & 63;
      uint32_t lo = f2bfu(hbuf[b * 128 + 2 * d]);
      uint32_t hi = f2bfu(hbuf[b * 128 + 2 * d + 1]);
      smem0[b * 192 + 128 + d] = lo | (hi << 16);
    }
    __syncthreads();
    v8f acc[2][2] = {};
    for (int k0 = 0; k0 < 384; k0 += 32) {
      int kp0 = k0 >> 1;
      v16bf a0 = a_frag(smem0 + (size_t)(0 + lm) * 192, kp0, hk);
      v16bf a1 = a_frag(smem0 + (size_t)(16 + lm) * 192, kp0, hk);
#pragma unroll
      for (int ni = 0; ni < 2; ++ni) {
        v16bf bb = b_frag(Wdw + (size_t)(n0 + ni * 16 + lm) * 192, kp0, hk);
        acc[0][ni] = wmma_bf16(a0, bb, acc[0][ni]);
        acc[1][ni] = wmma_bf16(a1, bb, acc[1][ni]);
      }
    }
    __syncthreads();
    bf16_t* G = (bf16_t*)smem0; // gates bf16 [32][512] (A region is dead now)
#pragma unroll
    for (int mi = 0; mi < 2; ++mi)
#pragma unroll
      for (int ni = 0; ni < 2; ++ni)
#pragma unroll
        for (int v = 0; v < 8; ++v)
          G[(mi * 16 + v + 8 * hk) * 512 + n0 + ni * 16 + lm] = f2bfu(acc[mi][ni][v]);
    __syncthreads();
    for (int e = tid; e < 4096; e += 512) { // cell update
      int b = e >> 7, j = e & 127;
      float gi = bf2f(G[b * 512 + j]) + Bs[j];
      float gf = bf2f(G[b * 512 + 128 + j]) + Bs[128 + j];
      float gg = bf2f(G[b * 512 + 256 + j]) + Bs[256 + j];
      float go = bf2f(G[b * 512 + 384 + j]) + Bs[384 + j];
      float c = sigmf(gf) * cbuf[e] + sigmf(gi) * tanhf(gg);
      float h = sigmf(go) * tanhf(c);
      cbuf[e] = c; hbuf[e] = h;
      if (encOut) encOut[((size_t)b * 256 + tt) * 256 + dir * 128 + j] = h;
      else        hsOut[((size_t)tt * 32 + b) * 256 + dir * 128 + j] = f2bfu(h);
    }
    __syncthreads();
  }
}

// ---------------- decoder attention + projections (1 block / step) ----------------
__global__ __launch_bounds__(1024) void k_attn(int t, int do_att,
    bf16_t* __restrict__ xatt, const bf16_t* __restrict__ Watt, const float* __restrict__ attb,
    const float* __restrict__ enc, const bf16_t* __restrict__ P, bf16_t* __restrict__ x1dst,
    const float* __restrict__ h2s,
    const float* __restrict__ melw, const float* __restrict__ melb,
    const float* __restrict__ gatew, const float* __restrict__ gateb,
    float* __restrict__ dout) {
  __shared__ float sbuf[32 * 256];
  __shared__ float pm[32 * 32];
  const int tid = threadIdx.x;
  const long GO = 1280000L, AW = 1296000L;
  if (t >= 1) { // mel/gate projections for step t-1 from h2
    int tm = t - 1;
    for (int idx = tid; idx < 32 * 80; idx += 1024) {
      int b = idx / 80, o = idx % 80;
      const float* h2 = h2s + (size_t)b * 1024;
      const float* wr = melw + (size_t)o * 1024;
      float s = melb[o];
      for (int j = 0; j < 1024; ++j) s += h2[j] * wr[j];
      dout[(size_t)b * 40000 + (size_t)tm * 80 + o] = s;
    }
    if (tid < 32) {
      const float* h2 = h2s + (size_t)tid * 1024;
      float s = gateb[0];
      for (int j = 0; j < 1024; ++j) s += h2[j] * gatew[j];
      dout[GO + (size_t)tid * 500 + tm] = s;
    }
  }
  if (!do_att) return;
  { // attention scores: [32,1280]bf16 @ attW[256][1280]^T  (32 waves, 1 tile each)
    const uint32_t* Xdw = (const uint32_t*)xatt;
    const uint32_t* Wdw = (const uint32_t*)Watt;
    int lid = tid & 31, w = tid >> 5;
    int lm = lid & 15, hk = lid >> 4;
    int m0 = (w & 1) * 16, n0 = (w >> 1) * 16;
    const uint32_t* arow = Xdw + (size_t)(m0 + lm) * 640;
    const uint32_t* brow = Wdw + (size_t)(n0 + lm) * 640;
    v8f acc = {};
    for (int k0 = 0; k0 < 1280; k0 += 32) {
      int kp0 = k0 >> 1;
      pf(brow + kp0 + 32);
      acc = wmma_bf16(a_frag(arow, kp0, hk), b_frag(brow, kp0, hk), acc);
    }
    int col = n0 + lm;
    float bv = attb[col];
#pragma unroll
    for (int v = 0; v < 8; ++v) sbuf[(m0 + v + 8 * hk) * 256 + col] = acc[v] + bv;
  }
  __syncthreads();
  { // row softmax (wave b handles batch row b)
    int b = tid >> 5, lane = tid & 31;
    float mx = -1e30f;
    for (int i = 0; i < 8; ++i) mx = fmaxf(mx, sbuf[b * 256 + lane + 32 * i]);
    pm[b * 32 + lane] = mx;
    __syncthreads();
    float rm = -1e30f;
    for (int i = 0; i < 32; ++i) rm = fmaxf(rm, pm[b * 32 + i]);
    float ls = 0.f;
    for (int i = 0; i < 8; ++i) {
      float e = __expf(sbuf[b * 256 + lane + 32 * i] - rm);
      sbuf[b * 256 + lane + 32 * i] = e;
      ls += e;
    }
    __syncthreads();
    pm[b * 32 + lane] = ls;
    __syncthreads();
    float sum = 0.f;
    for (int i = 0; i < 32; ++i) sum += pm[b * 32 + i];
    float inv = 1.f / sum;
    for (int i = 0; i < 8; ++i) {
      int n = lane + 32 * i;
      float v = sbuf[b * 256 + n] * inv;
      sbuf[b * 256 + n] = v;
      dout[AW + (size_t)b * 256 + n] = v; // final step's value survives
    }
  }
  __syncthreads();
  for (int e = tid; e < 8192; e += 1024) { // context = aw @ enc_out[b]
    int b = e >> 8, c = e & 255;
    const float* er = enc + ((size_t)b * 256) * 256 + c;
    const float* awr = sbuf + b * 256;
    float s = 0.f;
    for (int tau = 0; tau < 256; ++tau) s += awr[tau] * er[(size_t)tau * 256];
    xatt[(size_t)b * 1280 + 1024 + c] = f2bfu(s);   // for next step's attention input
    x1dst[(size_t)b * 1536 + 256 + c] = f2bfu(s);   // lstm1 input: ctx slot
  }
  { // copy prenet output p_t into lstm1 input (dword copy)
    const uint32_t* Pdw = (const uint32_t*)P;
    uint32_t* x1dw = (uint32_t*)x1dst;
    for (int e = tid; e < 4096; e += 1024) {
      int b = e >> 7, d = e & 127;
      x1dw[(size_t)b * 768 + d] = Pdw[((size_t)b * 500 + t) * 128 + d];
    }
  }
}

// ---------------- decoder LSTM cell: gates = X[32,K] @ Wcat[4096][K]^T, then update ----------------
__global__ __launch_bounds__(256) void k_cell(const bf16_t* __restrict__ X, int K,
                                              const bf16_t* __restrict__ W, const float* __restrict__ Bs,
                                              float* __restrict__ cs,
                                              bf16_t* __restrict__ out1, int p1, int o1,
                                              bf16_t* __restrict__ out2, int p2, int o2,
                                              float* __restrict__ hs) {
  __shared__ float gl[32 * 256]; // i|f|g|o x 64 cols for this block's hidden range
  const int tid = threadIdx.x, lid = tid & 31, w = tid >> 5;
  const int lm = lid & 15, hk = lid >> 4;
  const int j0 = blockIdx.x * 64;
  const int sec = w >> 1, half = w & 1;
  const int Kd = K >> 1;
  const uint32_t* Xdw = (const uint32_t*)X;
  const uint32_t* Wdw = (const uint32_t*)W;
  const uint32_t* arow0 = Xdw + (size_t)lm * Kd;
  const uint32_t* arow1 = Xdw + (size_t)(16 + lm) * Kd;
  v8f acc[2][2] = {};
  for (int k0 = 0; k0 < K; k0 += 32) {
    int kp0 = k0 >> 1;
    v16bf a0 = a_frag(arow0, kp0, hk);
    v16bf a1 = a_frag(arow1, kp0, hk);
#pragma unroll
    for (int ni = 0; ni < 2; ++ni) {
      int n = sec * 1024 + j0 + half * 32 + ni * 16 + lm;
      const uint32_t* wrow = Wdw + (size_t)n * Kd;
      v16bf bb = b_frag(wrow, kp0, hk);
      pf(wrow + kp0 + 32); // stream next weight k-chunk L2->L0
      acc[0][ni] = wmma_bf16(a0, bb, acc[0][ni]);
      acc[1][ni] = wmma_bf16(a1, bb, acc[1][ni]);
    }
  }
#pragma unroll
  for (int mi = 0; mi < 2; ++mi)
#pragma unroll
    for (int ni = 0; ni < 2; ++ni)
#pragma unroll
      for (int v = 0; v < 8; ++v)
        gl[(mi * 16 + v + 8 * hk) * 256 + sec * 64 + half * 32 + ni * 16 + lm] = acc[mi][ni][v];
  __syncthreads();
  for (int e = tid; e < 2048; e += 256) {
    int b = e >> 6, jl = e & 63, j = j0 + jl;
    float gi = gl[b * 256 + jl] + Bs[j];
    float gf = gl[b * 256 + 64 + jl] + Bs[1024 + j];
    float gg = gl[b * 256 + 128 + jl] + Bs[2048 + j];
    float go = gl[b * 256 + 192 + jl] + Bs[3072 + j];
    float c = sigmf(gf) * cs[b * 1024 + j] + sigmf(gi) * tanhf(gg);
    float h = sigmf(go) * tanhf(c);
    cs[b * 1024 + j] = c;
    out1[(size_t)b * p1 + o1 + j] = f2bfu(h);
    out2[(size_t)b * p2 + o2 + j] = f2bfu(h);
    if (hs) hs[b * 1024 + j] = h;
  }
}

// ---------------- host ----------------
static inline unsigned gsz(long n) {
  long b = (n + 2047) / 2048;
  if (b < 1) b = 1;
  return (unsigned)b;
}

extern "C" void kernel_launch(void* const* d_in, const int* in_sizes, int n_in,
                              void* d_out, int out_size, void* d_ws, size_t ws_size,
                              hipStream_t stream) {
  (void)in_sizes; (void)n_in; (void)out_size; (void)ws_size;
  const int*   text = (const int*)d_in[0];
  const float* mels = (const float*)d_in[2];
  const float* emb  = (const float*)d_in[3];
  const float* c1w = (const float*)d_in[4],  *c1b = (const float*)d_in[5];
  const float* c2w = (const float*)d_in[6],  *c2b = (const float*)d_in[7];
  const float* c3w = (const float*)d_in[8],  *c3b = (const float*)d_in[9];
  const float* ewih[4], *ewhh[4], *ebih[4], *ebhh[4];
  for (int i = 0; i < 4; ++i) {
    ewih[i] = (const float*)d_in[10 + 4 * i];
    ewhh[i] = (const float*)d_in[11 + 4 * i];
    ebih[i] = (const float*)d_in[12 + 4 * i];
    ebhh[i] = (const float*)d_in[13 + 4 * i];
  }
  const float* p0w = (const float*)d_in[26], *p0b = (const float*)d_in[27];
  const float* p1w = (const float*)d_in[28], *p1b = (const float*)d_in[29];
  const float* d1wih = (const float*)d_in[30], *d1whh = (const float*)d_in[31];
  const float* d1bih = (const float*)d_in[32], *d1bhh = (const float*)d_in[33];
  const float* d2wih = (const float*)d_in[34], *d2whh = (const float*)d_in[35];
  const float* d2bih = (const float*)d_in[36], *d2bhh = (const float*)d_in[37];
  const float* attw = (const float*)d_in[38], *attb = (const float*)d_in[39];
  const float* melw = (const float*)d_in[40], *melb = (const float*)d_in[41];
  const float* gatew = (const float*)d_in[42], *gateb = (const float*)d_in[43];

  char* ws = (char*)d_ws;
  size_t off = 0;
  auto alloc = [&](size_t bytes) -> void* {
    void* p = ws + off;
    off = (off + bytes + 255) & ~((size_t)255);
    return p;
  };
  bf16_t* Wc1 = (bf16_t*)alloc(256L * 2560 * 2);
  bf16_t* Wc2 = (bf16_t*)alloc(256L * 1280 * 2);
  bf16_t* Wc3 = (bf16_t*)alloc(256L * 1280 * 2);
  bf16_t* Wenc[4]; float* Benc[4];
  for (int i = 0; i < 4; ++i) { Wenc[i] = (bf16_t*)alloc(512L * 384 * 2); Benc[i] = (float*)alloc(512L * 4); }
  bf16_t* Wp0 = (bf16_t*)alloc(256L * 96 * 2);
  bf16_t* Wp1 = (bf16_t*)alloc(256L * 256 * 2);
  bf16_t* Watt = (bf16_t*)alloc(256L * 1280 * 2);
  bf16_t* W1 = (bf16_t*)alloc(4096L * 1536 * 2);
  float*  B1 = (float*)alloc(4096L * 4);
  bf16_t* W2 = (bf16_t*)alloc(4096L * 2048 * 2);
  float*  B2 = (float*)alloc(4096L * 4);
  bf16_t* xpad0 = (bf16_t*)alloc(32L * 260 * 512 * 2);
  bf16_t* xpad1 = (bf16_t*)alloc(32L * 260 * 256 * 2);
  bf16_t* xpad2 = (bf16_t*)alloc(32L * 260 * 256 * 2);
  bf16_t* c3o   = (bf16_t*)alloc(32L * 256 * 256 * 2);
  bf16_t* hs0   = (bf16_t*)alloc(256L * 32 * 256 * 2);
  float*  encO  = (float*)alloc(32L * 256 * 256 * 4);
  bf16_t* melsb = (bf16_t*)alloc(16000L * 96 * 2);
  bf16_t* P1b   = (bf16_t*)alloc(16000L * 256 * 2);
  bf16_t* Pbuf  = (bf16_t*)alloc(16000L * 256 * 2);
  bf16_t* x1b   = (bf16_t*)alloc(2L * 32 * 1536 * 2);
  bf16_t* x2b   = (bf16_t*)alloc(2L * 32 * 2048 * 2);
  bf16_t* xatt  = (bf16_t*)alloc(32L * 1280 * 2);
  float*  c1s = (float*)alloc(32L * 1024 * 4);
  float*  c2s = (float*)alloc(32L * 1024 * 4);
  float*  h2s = (float*)alloc(32L * 1024 * 4);
  float*  dout = (float*)d_out;

  // ---- init / weight prep ----
  k_zero<<<gsz(1064960), 256, 0, stream>>>((uint32_t*)xpad1, 1064960);
  k_zero<<<gsz(1064960), 256, 0, stream>>>((uint32_t*)xpad2, 1064960);
  k_zero<<<gsz(49152), 256, 0, stream>>>((uint32_t*)x1b, 49152);
  k_zero<<<gsz(65536), 256, 0, stream>>>((uint32_t*)x2b, 65536);
  k_zero<<<gsz(20480), 256, 0, stream>>>((uint32_t*)xatt, 20480);
  k_zero<<<gsz(32768), 256, 0, stream>>>((uint32_t*)c1s, 32768);
  k_zero<<<gsz(32768), 256, 0, stream>>>((uint32_t*)c2s, 32768);
  k_zero<<<gsz(32768), 256, 0, stream>>>((uint32_t*)h2s, 32768);

  k_prep_conv<<<gsz(256L * 2560), 256, 0, stream>>>(c1w, 512, Wc1);
  k_prep_conv<<<gsz(256L * 1280), 256, 0, stream>>>(c2w, 256, Wc2);
  k_prep_conv<<<gsz(256L * 1280), 256, 0, stream>>>(c3w, 256, Wc3);
  for (int i = 0; i < 4; ++i) {
    k_prep_cat<<<gsz(512L * 384), 256, 0, stream>>>(ewih[i], 256, ewhh[i], 128, Wenc[i], 512);
    k_addb<<<gsz(512), 256, 0, stream>>>(ebih[i], ebhh[i], Benc[i], 512);
  }
  k_prep_cat<<<gsz(256L * 96), 256, 0, stream>>>(p0w, 80, (const float*)nullptr, 16, Wp0, 256);
  k_prep_cat<<<gsz(256L * 256), 256, 0, stream>>>(p1w, 256, (const float*)nullptr, 0, Wp1, 256);
  k_prep_cat<<<gsz(256L * 1280), 256, 0, stream>>>(attw, 1280, (const float*)nullptr, 0, Watt, 256);
  k_prep_cat<<<gsz(4096L * 1536), 256, 0, stream>>>(d1wih, 512, d1whh, 1024, W1, 4096);
  k_addb<<<gsz(4096), 256, 0, stream>>>(d1bih, d1bhh, B1, 4096);
  k_prep_cat<<<gsz(4096L * 2048), 256, 0, stream>>>(d2wih, 1024, d2whh, 1024, W2, 4096);
  k_addb<<<gsz(4096), 256, 0, stream>>>(d2bih, d2bhh, B2, 4096);
  k_prep_cat<<<gsz(16000L * 96), 256, 0, stream>>>(mels, 80, (const float*)nullptr, 16, melsb, 16000);

  // ---- encoder ----
  k_embed<<<8320, 256, 0, stream>>>(text, emb, xpad0);
  k_convgemm<<<256, 256, 0, stream>>>(xpad0, 512, Wc1, c1b, xpad1, 260, 2);
  k_convgemm<<<256, 256, 0, stream>>>(xpad1, 256, Wc2, c2b, xpad2, 260, 2);
  k_convgemm<<<256, 256, 0, stream>>>(xpad2, 256, Wc3, c3b, c3o, 256, 0);
  k_enc<<<2, 512, 0, stream>>>(c3o, 0, Wenc[0], Benc[0], Wenc[1], Benc[1], hs0, (float*)nullptr);
  k_enc<<<2, 512, 0, stream>>>(hs0, 1, Wenc[2], Benc[2], Wenc[3], Benc[3], (bf16_t*)nullptr, encO);

  // ---- prenet (all 500 steps at once) ----
  k_gemm_lin<<<500, 256, 0, stream>>>(melsb, Wp0, p0b, P1b, 16000, 256, 96);
  k_gemm_lin<<<500, 256, 0, stream>>>(P1b, Wp1, p1b, Pbuf, 16000, 256, 256);

  // ---- decoder: 500 sequential steps ----
  for (int t = 0; t < 500; ++t) {
    bf16_t* x1c = x1b + (size_t)(t & 1) * 32 * 1536;
    bf16_t* x1n = x1b + (size_t)((t + 1) & 1) * 32 * 1536;
    bf16_t* x2c = x2b + (size_t)(t & 1) * 32 * 2048;
    bf16_t* x2n = x2b + (size_t)((t + 1) & 1) * 32 * 2048;
    k_attn<<<1, 1024, 0, stream>>>(t, 1, xatt, Watt, attb, encO, Pbuf, x1c, h2s,
                                   melw, melb, gatew, gateb, dout);
    k_cell<<<16, 256, 0, stream>>>(x1c, 1536, W1, B1, c1s, x1n, 1536, 512, x2c, 2048, 0, (float*)nullptr);
    k_cell<<<16, 256, 0, stream>>>(x2c, 2048, W2, B2, c2s, x2n, 2048, 1024, xatt, 1280, 0, h2s);
  }
  // final mel/gate projection for t=499
  k_attn<<<1, 1024, 0, stream>>>(500, 0, xatt, Watt, attb, encO, Pbuf, x1b, h2s,
                                 melw, melb, gatew, gateb, dout);
}
